// CustomDeepSpatialAutoencoder_47828755808305
// MI455X (gfx1250) — matrix operand order repaired and checked by reading, not compiled, not run on hardware
//
#include <hip/hip_runtime.h>
#include <math.h>

// ---------------------------------------------------------------------------
// CDNA5 (gfx1250, wave32) implicit-GEMM autoencoder forward.
// All convs / convtranspose / linear run through one templated WMMA kernel
// (v_wmma_f32_16x16x32_f16), f32 accumulate, f16 operands.
// Block = 256 threads (8 waves) split MT x NT; each wave computes a 32x16
// output patch (two 16x16 WMMA tiles) from one shared LDS im2col B-tile.
// BN = deterministic two-pass partial-sum reduction (no float atomics).
// ---------------------------------------------------------------------------

typedef __attribute__((ext_vector_type(16))) _Float16 v16h;
typedef __attribute__((ext_vector_type(8)))  _Float16 v8h;
typedef __attribute__((ext_vector_type(4)))  _Float16 v4h;
typedef __attribute__((ext_vector_type(2)))  _Float16 v2h;
typedef __attribute__((ext_vector_type(8)))  float    v8f;

#define NEG_SLOPE 0.01f
#define BN_EPS    1e-5f

#define MODE_CONV   0   // stride-s conv, pad 1, 3x3 (stride 1 or 2)
#define MODE_CONVT  1   // convtranspose k3 s2 p1 op1 (flipped-weight formulation)
#define MODE_DENSE  2   // plain GEMM (linear layer)

#define ACT_NONE  0
#define ACT_LEAKY 1
#define ACT_TANH  2

// ---------------------------------------------------------------------------
// Weight packing: fp32 -> f16, [Co_pad][K_pad] row-major, zero padded.
// transposed==1: torch ConvTranspose weight (Ci,Co,kh,kw), spatially flipped.
// ---------------------------------------------------------------------------
__global__ void pack_weights(const float* __restrict__ w, _Float16* __restrict__ out,
                             int Co, int Ci, int ksz, int CoPad, int Kpad, int transposed)
{
    int i = blockIdx.x * blockDim.x + threadIdx.x;
    if (i >= CoPad * Kpad) return;
    int o = i / Kpad;
    int k = i - o * Kpad;
    float v = 0.0f;
    int K = Ci * ksz;
    if (o < Co && k < K) {
        int ci = k / ksz;
        int r  = k - ci * ksz;
        if (!transposed) v = w[(o * Ci + ci) * ksz + r];               // OIHW
        else             v = w[(ci * Co + o) * ksz + (ksz - 1 - r)];   // IOHW, flip HW
    }
    out[i] = (_Float16)v;
}

// ---------------------------------------------------------------------------
// Unified implicit-GEMM WMMA kernel, templated on mode and wave tiling.
// MT m-groups (32 channels each) x NT n-slices (16 px each), MT*NT == 8.
// LDS B tile: (NT*16 pixels) x 32 K halves, rebuilt each K-step.
// ---------------------------------------------------------------------------
template<int MODE, int MT, int NT>
__global__ __launch_bounds__(256)
void igemm_wmma(const float* __restrict__ in, const _Float16* __restrict__ wpk,
                const float* __restrict__ bias, float* __restrict__ out,
                int act,
                int B, int Ci, int Hi, int Wi,
                int Co, int Ho, int Wo,
                int Kreal, int Kpad, int stride)
{
    constexpr int ELEMS = NT * 2;          // halves per thread per K-step
    constexpr int NPB   = NT * 16;         // pixels per block

    const int tid  = threadIdx.x;
    const int wave = tid >> 5;
    const int lane = tid & 31;
    const int ni   = wave % NT;            // n-slice of this wave
    const int mi   = wave / NT;            // m-group of this wave
    const int HW   = Ho * Wo;
    const int Npix = B * HW;
    const int pixBase = blockIdx.x * NPB;

    __shared__ __align__(32) _Float16 ldsB[NPB * 32];

    // ---- fill assignment: ELEMS consecutive k-halves of one pixel per thread
    const int lin0  = tid * ELEMS;
    const int fpix  = lin0 >> 5;           // pixel slot 0..NPB-1
    const int fk0   = lin0 & 31;           // k offset within the 32-wide step
    const int pixF  = pixBase + fpix;
    const bool fvalid = pixF < Npix;
    int nF = 0, oyF = 0, oxF = 0;
    if (fvalid) {
        nF = pixF / HW;
        int sp = pixF - nF * HW;
        oyF = sp / Wo;
        oxF = sp - oyF * Wo;
    }
    const float* __restrict__ inN = in + (size_t)nF * Ci * Hi * Wi;
    _Float16* __restrict__ ldst = &ldsB[fpix * 32 + fk0];

    // ---- A fragment pointers (two m-tiles per wave) ----
    const int mtileG = (blockIdx.y * MT + mi) * 2;            // in 16-row tiles
    const int mrow0  = mtileG * 16 + (lane & 15);
    const int lsub   = (lane & 16) ? 16 : 0;                  // A-layout K sub-offset
    const _Float16* __restrict__ wr0 = wpk + (size_t)mrow0 * Kpad + lsub;
    const _Float16* __restrict__ wr1 = wr0 + (size_t)16 * Kpad;
    const _Float16* __restrict__ bb  =
        &ldsB[(ni * 16 + (lane & 15)) * 32 + ((lane & 16) ? 16 : 0)];

    v8f acc0 = {}, acc1 = {};

    for (int kstep = 0; kstep < Kpad; kstep += 32) {
        // ================= cooperative im2col fill (vectorized LDS store) ====
        _Float16 tmp[ELEMS];
        const int k0 = kstep + fk0;
        if (!fvalid) {
            #pragma unroll
            for (int s = 0; s < ELEMS; ++s) tmp[s] = (_Float16)0.0f;
        } else if constexpr (MODE == MODE_DENSE) {
            #pragma unroll
            for (int s = 0; s < ELEMS; ++s) {
                int k = k0 + s;
                float v = (k < Kreal) ? in[pixF * Kreal + k] : 0.0f;
                tmp[s] = (_Float16)v;
            }
        } else {
            // decompose once per K-step, then carry (ci,ky,kx) incrementally
            int ci = k0 / 9;
            int r  = k0 - ci * 9;
            int ky = r / 3;
            int kx = r - ky * 3;
            #pragma unroll
            for (int s = 0; s < ELEMS; ++s) {
                float v = 0.0f;
                if ((k0 + s) < Kreal) {
                    if constexpr (MODE == MODE_CONV) {
                        int iy = oyF * stride + ky - 1;
                        int ix = oxF * stride + kx - 1;
                        if (iy >= 0 && iy < Hi && ix >= 0 && ix < Wi)
                            v = inN[(ci * Hi + iy) * Wi + ix];
                    } else { // MODE_CONVT: lhs_dilation=2, pad (1,2)
                        int ty = oyF + ky - 1;
                        int tx = oxF + kx - 1;
                        if (((ty | tx) >= 0) && !((ty | tx) & 1)) {
                            int iy = ty >> 1, ix = tx >> 1;
                            if (iy < Hi && ix < Wi)
                                v = inN[(ci * Hi + iy) * Wi + ix];
                        }
                    }
                }
                tmp[s] = (_Float16)v;
                if (++kx == 3) { kx = 0; if (++ky == 3) { ky = 0; ++ci; } }
            }
        }
        // one vector DS store per thread (fk0 is a multiple of ELEMS -> aligned)
        if constexpr (ELEMS == 2) {
            v2h t; t[0] = tmp[0]; t[1] = tmp[1];
            *(v2h*)ldst = t;
        } else if constexpr (ELEMS == 4) {
            v4h t;
            #pragma unroll
            for (int s = 0; s < 4; ++s) t[s] = tmp[s];
            *(v4h*)ldst = t;
        } else if constexpr (ELEMS == 8) {
            v8h t;
            #pragma unroll
            for (int s = 0; s < 8; ++s) t[s] = tmp[s];
            *(v8h*)ldst = t;
        } else {
            v8h t0, t1;
            #pragma unroll
            for (int s = 0; s < 8; ++s) { t0[s] = tmp[s]; t1[s] = tmp[s + 8]; }
            *(v8h*)ldst = t0;
            *((v8h*)ldst + 1) = t1;
        }
        __syncthreads();

        // ================= fragments + 2x WMMA =================
        const _Float16* w0 = wr0 + kstep;
        const _Float16* w1 = wr1 + kstep;
        __builtin_prefetch(w0 + 32, 0, 1);   // next K-step weight row (global_prefetch_b8)

        v8h a0l = *(const v8h*)(w0);
        v8h a0h = *(const v8h*)(w0 + 8);
        v8h a1l = *(const v8h*)(w1);
        v8h a1h = *(const v8h*)(w1 + 8);
        v16h A0, A1;
        #pragma unroll
        for (int i = 0; i < 8; ++i) {
            A0[i] = a0l[i]; A0[i + 8] = a0h[i];
            A1[i] = a1l[i]; A1[i + 8] = a1h[i];
        }
        v16h Bf = *(const v16h*)bb;

        acc0 = __builtin_amdgcn_wmma_f32_16x16x32_f16(
                   false, A0, false, Bf, (short)0, acc0, false, false);
        acc1 = __builtin_amdgcn_wmma_f32_16x16x32_f16(
                   false, A1, false, Bf, (short)0, acc1, false, false);
        __syncthreads();
    }

    // ================= epilogue: bias + activation + store =================
    const int pix = pixBase + ni * 16 + (lane & 15);
    if (pix < Npix) {
        const int n  = pix / HW;
        const int sp = pix - n * HW;
        const int mb0 = mtileG * 16 + ((lane & 16) ? 8 : 0);
        #pragma unroll
        for (int r = 0; r < 16; ++r) {
            int m = (r < 8) ? (mb0 + r) : (mb0 + 8 + r);   // tile0: mb0+r, tile1: mb0+16+(r-8)
            if (m < Co) {
                float v = (r < 8) ? acc0[r] : acc1[r - 8];
                if (bias) v += bias[m];
                if (act == ACT_LEAKY)     v = (v > 0.0f) ? v : NEG_SLOPE * v;
                else if (act == ACT_TANH) v = tanhf(v);
                if constexpr (MODE == MODE_DENSE) out[pix * Co + m] = v;
                else                              out[(n * Co + m) * HW + sp] = v;
            }
        }
    }
}

// ---------------------------------------------------------------------------
// BatchNorm (training stats) — deterministic two-pass reduction.
// ---------------------------------------------------------------------------
#define BN_CHUNKS 64

__global__ void bn_stats(const float* __restrict__ x, float* __restrict__ part,
                         int B, int C, int HW)
{
    const int c     = blockIdx.x;
    const int chunk = blockIdx.y;
    const int total = B * HW;
    float s = 0.0f, s2 = 0.0f;
    for (int j = chunk * blockDim.x + threadIdx.x; j < total; j += BN_CHUNKS * blockDim.x) {
        int n  = j / HW;
        int sp = j - n * HW;
        float v = x[(n * C + c) * HW + sp];
        s += v; s2 += v * v;
    }
    __shared__ float rs[256], rq[256];
    rs[threadIdx.x] = s; rq[threadIdx.x] = s2;
    __syncthreads();
    for (int off = 128; off > 0; off >>= 1) {
        if (threadIdx.x < off) {
            rs[threadIdx.x] += rs[threadIdx.x + off];
            rq[threadIdx.x] += rq[threadIdx.x + off];
        }
        __syncthreads();
    }
    if (threadIdx.x == 0) {
        part[(c * BN_CHUNKS + chunk) * 2 + 0] = rs[0];
        part[(c * BN_CHUNKS + chunk) * 2 + 1] = rq[0];
    }
}

__global__ void bn_reduce(const float* __restrict__ part, float* __restrict__ stats, int C)
{
    int c = blockIdx.x * blockDim.x + threadIdx.x;
    if (c >= C) return;
    float s = 0.0f, s2 = 0.0f;
    for (int k = 0; k < BN_CHUNKS; ++k) {        // fixed order -> deterministic
        s  += part[(c * BN_CHUNKS + k) * 2 + 0];
        s2 += part[(c * BN_CHUNKS + k) * 2 + 1];
    }
    stats[c]     = s;
    stats[C + c] = s2;
}

__global__ void bn_apply(float* __restrict__ x, const float* __restrict__ stats,
                         const float* __restrict__ g, const float* __restrict__ b,
                         int C, int HW, int total, float invN)
{
    int i = blockIdx.x * blockDim.x + threadIdx.x;
    if (i >= total) return;
    int c = (i / HW) % C;
    float m   = stats[c] * invN;
    float var = stats[C + c] * invN - m * m;
    float v = (x[i] - m) * rsqrtf(var + BN_EPS);
    v = g[c] * v + b[c];
    x[i] = (v > 0.0f) ? v : NEG_SLOPE * v;
}

// ---------------------------------------------------------------------------
// Keypoint gather: z[b][c*2+j] = e4[b][c][yi*6+xi]
// ---------------------------------------------------------------------------
__global__ void gather_kp(const float* __restrict__ e4, const int* __restrict__ p,
                          float* __restrict__ z, int B)
{
    int i = blockIdx.x * blockDim.x + threadIdx.x;
    if (i >= B * 64) return;
    int b = i >> 6;
    int t = i & 63;
    int c = t >> 1;
    int j = t & 1;
    int px = p[(b * 2 + j) * 2 + 0];
    int py = p[(b * 2 + j) * 2 + 1];
    // pix = ((p/95*2-1)+1)*(6-1)/2 = p*5/95, truncated to int (same as reference)
    int xi = (int)((float)px * (5.0f / 95.0f));
    int yi = (int)((float)py * (5.0f / 95.0f));
    int idx = yi * 6 + xi;
    z[b * 64 + c * 2 + j] = e4[(b * 32 + c) * 36 + idx];
}

// ---------------------------------------------------------------------------
// Host orchestration
// ---------------------------------------------------------------------------
extern "C" void kernel_launch(void* const* d_in, const int* in_sizes, int n_in,
                              void* d_out, int out_size, void* d_ws, size_t ws_size,
                              hipStream_t stream)
{
    (void)in_sizes; (void)n_in; (void)out_size; (void)ws_size;

    const float* x = (const float*)d_in[0];
    const int*   p = (const int*)d_in[1];
    // params in setup_inputs() insertion order
    const float* ew[4]  = {(const float*)d_in[2],  (const float*)d_in[6],
                           (const float*)d_in[10], (const float*)d_in[14]};
    const float* ebi[4] = {(const float*)d_in[3],  (const float*)d_in[7],
                           (const float*)d_in[11], (const float*)d_in[15]};
    const float* eg[4]  = {(const float*)d_in[4],  (const float*)d_in[8],
                           (const float*)d_in[12], (const float*)d_in[16]};
    const float* ebt[4] = {(const float*)d_in[5],  (const float*)d_in[9],
                           (const float*)d_in[13], (const float*)d_in[17]};
    const float* lw = (const float*)d_in[18];
    const float* lb = (const float*)d_in[19];
    const float* dw[3]  = {(const float*)d_in[20], (const float*)d_in[24], (const float*)d_in[28]};
    const float* db[3]  = {(const float*)d_in[21], (const float*)d_in[25], (const float*)d_in[29]};
    const float* dg[3]  = {(const float*)d_in[22], (const float*)d_in[26], (const float*)d_in[30]};
    const float* dbt[3] = {(const float*)d_in[23], (const float*)d_in[27], (const float*)d_in[31]};
    const float* fw  = (const float*)d_in[32];
    const float* fb  = (const float*)d_in[33];
    const float* fg  = (const float*)d_in[34];
    const float* fbt = (const float*)d_in[35];
    const float* cw  = (const float*)d_in[36];
    const float* cb  = (const float*)d_in[37];

    // ---- workspace carve-out (256B aligned) ----
    char* wsb = (char*)d_ws;
    size_t off = 0;
    auto alloc = [&](size_t bytes) -> void* {
        off = (off + 255) & ~(size_t)255;
        void* r = wsb + off;
        off += bytes;
        return r;
    };
    float* bufA  = (float*)alloc((size_t)256 * 32 * 96 * 96 * sizeof(float)); // 302 MB (max A)
    float* bufB  = (float*)alloc((size_t)256 * 32 * 48 * 48 * sizeof(float)); //  75 MB (max B)
    float* zbuf  = (float*)alloc((size_t)256 * 64 * sizeof(float));
    float* stats = (float*)alloc((size_t)2 * 128 * sizeof(float));
    float* part  = (float*)alloc((size_t)128 * BN_CHUNKS * 2 * sizeof(float));

    _Float16* pw_e0  = (_Float16*)alloc((size_t)32   * 32   * 2);
    _Float16* pw_e1  = (_Float16*)alloc((size_t)64   * 288  * 2);
    _Float16* pw_e2  = (_Float16*)alloc((size_t)128  * 576  * 2);
    _Float16* pw_e3  = (_Float16*)alloc((size_t)32   * 1152 * 2);
    _Float16* pw_lin = (_Float16*)alloc((size_t)1152 * 64   * 2);
    _Float16* pw_d0  = (_Float16*)alloc((size_t)128  * 288  * 2);
    _Float16* pw_d1  = (_Float16*)alloc((size_t)64   * 1152 * 2);
    _Float16* pw_d2  = (_Float16*)alloc((size_t)32   * 576  * 2);
    _Float16* pw_f   = (_Float16*)alloc((size_t)32   * 288  * 2);
    _Float16* pw_c   = (_Float16*)alloc((size_t)32   * 288  * 2);   // Co=3 padded to 32

    auto pack = [&](const float* w, _Float16* o, int Co, int Ci, int ksz,
                    int CoPad, int Kpad, int tr) {
        int tot = CoPad * Kpad;
        pack_weights<<<(tot + 255) / 256, 256, 0, stream>>>(w, o, Co, Ci, ksz, CoPad, Kpad, tr);
    };
    pack(ew[0], pw_e0,   32,   3, 9,   32,   32, 0);
    pack(ew[1], pw_e1,   64,  32, 9,   64,  288, 0);
    pack(ew[2], pw_e2,  128,  64, 9,  128,  576, 0);
    pack(ew[3], pw_e3,   32, 128, 9,   32, 1152, 0);
    pack(lw,    pw_lin, 1152, 64, 1, 1152,   64, 0);
    pack(dw[0], pw_d0,  128,  32, 9,  128,  288, 1);
    pack(dw[1], pw_d1,   64, 128, 9,   64, 1152, 1);
    pack(dw[2], pw_d2,   32,  64, 9,   32,  576, 1);
    pack(fw,    pw_f,    32,  32, 9,   32,  288, 1);
    pack(cw,    pw_c,     3,  32, 9,   32,  288, 0);

    // MT = m-groups (32 channels each) per block, NT = 8/MT n-slices (16 px each)
    auto gemm = [&](const float* in, const _Float16* wpk, const float* bias, float* out,
                    int mode, int act, int B, int Ci, int Hi, int Wi,
                    int Co, int Ho, int Wo, int Kreal, int Kpad, int stride, int CoPad) {
        int MT = CoPad / 32; if (MT > 4) MT = 4;
        int NT = 8 / MT;
        int Npix = B * Ho * Wo;
        dim3 g((Npix + NT * 16 - 1) / (NT * 16), CoPad / (MT * 32));
#define IG_LAUNCH(M_, MT_, NT_)                                                     \
        igemm_wmma<M_, MT_, NT_><<<g, 256, 0, stream>>>(in, wpk, bias, out, act,    \
            B, Ci, Hi, Wi, Co, Ho, Wo, Kreal, Kpad, stride)
        if (mode == MODE_CONV) {
            if (MT == 1)      IG_LAUNCH(MODE_CONV, 1, 8);
            else if (MT == 2) IG_LAUNCH(MODE_CONV, 2, 4);
            else              IG_LAUNCH(MODE_CONV, 4, 2);
        } else if (mode == MODE_CONVT) {
            if (MT == 1)      IG_LAUNCH(MODE_CONVT, 1, 8);
            else if (MT == 2) IG_LAUNCH(MODE_CONVT, 2, 4);
            else              IG_LAUNCH(MODE_CONVT, 4, 2);
        } else {
            IG_LAUNCH(MODE_DENSE, 4, 2);
        }
#undef IG_LAUNCH
    };

    auto bnrun = [&](float* act, const float* g, const float* b, int B, int C, int HW) {
        dim3 gs(C, BN_CHUNKS);
        bn_stats<<<gs, 256, 0, stream>>>(act, part, B, C, HW);
        bn_reduce<<<(C + 255) / 256, 256, 0, stream>>>(part, stats, C);
        int tot = B * C * HW;
        bn_apply<<<(tot + 255) / 256, 256, 0, stream>>>(act, stats, g, b, C, HW, tot,
                                                        1.0f / (float)(B * HW));
    };

    const int B = 256;

    // ---- encoder ----
    gemm(x,    pw_e0, ebi[0], bufA, MODE_CONV, ACT_NONE, B,   3, 96, 96,  32, 48, 48,   27,   32, 2,  32);
    bnrun(bufA, eg[0], ebt[0], B,  32, 48 * 48);
    gemm(bufA, pw_e1, ebi[1], bufB, MODE_CONV, ACT_NONE, B,  32, 48, 48,  64, 24, 24,  288,  288, 2,  64);
    bnrun(bufB, eg[1], ebt[1], B,  64, 24 * 24);
    gemm(bufB, pw_e2, ebi[2], bufA, MODE_CONV, ACT_NONE, B,  64, 24, 24, 128, 12, 12,  576,  576, 2, 128);
    bnrun(bufA, eg[2], ebt[2], B, 128, 12 * 12);
    gemm(bufA, pw_e3, ebi[3], bufB, MODE_CONV, ACT_NONE, B, 128, 12, 12,  32,  6,  6, 1152, 1152, 2,  32);
    bnrun(bufB, eg[3], ebt[3], B,  32, 6 * 6);

    // ---- keypoint gather: bufB (B,32,6,6) -> z (B,64) ----
    gather_kp<<<(B * 64 + 255) / 256, 256, 0, stream>>>(bufB, p, zbuf, B);

    // ---- linear 64 -> 1152, bias + leaky; output == NCHW (B,32,6,6) ----
    gemm(zbuf, pw_lin, lb, bufA, MODE_DENSE, ACT_LEAKY, B, 64, 1, 1, 1152, 1, 1, 64, 64, 1, 1152);

    // ---- decoder (convtranspose) ----
    gemm(bufA, pw_d0, db[0], bufB, MODE_CONVT, ACT_NONE, B,  32,  6,  6, 128, 12, 12,  288,  288, 1, 128);
    bnrun(bufB, dg[0], dbt[0], B, 128, 12 * 12);
    gemm(bufB, pw_d1, db[1], bufA, MODE_CONVT, ACT_NONE, B, 128, 12, 12,  64, 24, 24, 1152, 1152, 1,  64);
    bnrun(bufA, dg[1], dbt[1], B,  64, 24 * 24);
    gemm(bufA, pw_d2, db[2], bufB, MODE_CONVT, ACT_NONE, B,  64, 24, 24,  32, 48, 48,  576,  576, 1,  32);
    bnrun(bufB, dg[2], dbt[2], B,  32, 48 * 48);
    gemm(bufB, pw_f,  fb,    bufA, MODE_CONVT, ACT_NONE, B,  32, 48, 48,  32, 96, 96,  288,  288, 1,  32);
    bnrun(bufA, fg, fbt, B, 32, 96 * 96);

    // ---- final conv s1 p1 (32->3) + tanh -> d_out ----
    gemm(bufA, pw_c, cb, (float*)d_out, MODE_CONV, ACT_TANH, B, 32, 96, 96, 3, 96, 96, 288, 288, 1, 32);
}
